// KGEModule_9637906612511
// MI455X (gfx1250) — matrix operand order repaired and checked by reading, not compiled, not run on hardware
//
#include <hip/hip_runtime.h>
#include <math.h>

// ---- problem constants (match reference) ----
#define N_ENT  50000
#define NDIM   256
#define BB     1024
#define NTILES (N_ENT / 16)   // 3125, exact
#define ESPLIT 8              // entity-range splits per pass (grid.z)

typedef __attribute__((ext_vector_type(16))) _Float16 v16h;
typedef __attribute__((ext_vector_type(8)))  _Float16 v8h;
typedef __attribute__((ext_vector_type(4)))  _Float16 v4h;
typedef __attribute__((ext_vector_type(8)))  float    v8f;

__device__ __forceinline__ v16h cat8(v8h a, v8h b) {
    return __builtin_shufflevector(a, b, 0,1,2,3,4,5,6,7,8,9,10,11,12,13,14,15);
}

// Load one 16-entity x 256-deep B tile (this lane's share): 8 fragments.
__device__ __forceinline__ void loadB(const _Float16* __restrict__ base, v16h B[8]) {
    #pragma unroll
    for (int kc = 0; kc < 8; ++kc) {
        v8h b0 = *(const v8h*)(base + kc * 32);
        v8h b1 = *(const v8h*)(base + kc * 32 + 8);
        B[kc] = cat8(b0, b1);
    }
}

// Full-K (256) 16x16 logits tile: 8 chained WMMAs.
__device__ __forceinline__ v8f mmTile(const v16h A[8], const v16h B[8]) {
    v8f acc = {};
    #pragma unroll
    for (int kc = 0; kc < 8; ++kc) {
        acc = __builtin_amdgcn_wmma_f32_16x16x32_f16(
            false, A[kc], false, B[kc], (short)0, acc, false, false);
    }
    return acc;
}

// Per-lane online softmax update (no cross-lane traffic in the loop).
__device__ __forceinline__ void smUpdate(const v8f& acc, float mr[8], float sr[8]) {
    #pragma unroll
    for (int g = 0; g < 8; ++g) {
        float x  = acc[g];
        float mo = mr[g];
        float mn = fmaxf(mo, x);
        sr[g] = sr[g] * __expf(mo - mn) + __expf(x - mn);
        mr[g] = mn;
    }
}

// ------------------------------------------------------------------
// Kernel 0: zero the scalar accumulator
// ------------------------------------------------------------------
__global__ void k_init(float* accum) { accum[0] = 0.0f; }

// ------------------------------------------------------------------
// Kernel 1: BN stats (mean, invstd) over the three gathered batches.
// ------------------------------------------------------------------
__global__ void k_bnstats(const int* __restrict__ facts,
                          const float* __restrict__ ent_w,
                          const float* __restrict__ rel_w,
                          float* __restrict__ stats) {
    const int d = threadIdx.x;
    const int t = blockIdx.x;                   // 0: ent[h], 1: ent[t], 2: rel[r]
    const float* tab = (t == 2) ? rel_w : ent_w;
    float s = 0.0f, q = 0.0f;
    for (int b = 0; b < BB; ++b) {
        int row = facts[3 * b + t];
        float x = tab[row * NDIM + d];
        s += x; q += x * x;
    }
    float mean = s * (1.0f / BB);
    float var  = q * (1.0f / BB) - mean * mean;
    stats[t * 512 + d]       = mean;
    stats[t * 512 + 256 + d] = rsqrtf(var + 1e-5f);
}

// ------------------------------------------------------------------
// Kernel 2: fused BN-normalize + alpha einsum -> head/tail vecs (f16)
// ------------------------------------------------------------------
__global__ void k_build(const int* __restrict__ facts,
                        const int* __restrict__ arch,
                        const float* __restrict__ ent_w,
                        const float* __restrict__ rel_w,
                        const float* __restrict__ bneg, const float* __restrict__ bneb,
                        const float* __restrict__ bnrg, const float* __restrict__ bnrb,
                        const float* __restrict__ stats,
                        _Float16* __restrict__ hvec,
                        _Float16* __restrict__ tvec) {
    __shared__ float he[NDIM], te[NDIM], re[NDIM], al[64];
    const int b = blockIdx.x, d = threadIdx.x;
    const int hI = facts[3 * b], tI = facts[3 * b + 1], rI = facts[3 * b + 2];
    he[d] = (ent_w[hI * NDIM + d] - stats[d])        * stats[256 + d]  * bneg[d] + bneb[d];
    te[d] = (ent_w[tI * NDIM + d] - stats[512 + d])  * stats[768 + d]  * bneg[d] + bneb[d];
    re[d] = (rel_w[rI * NDIM + d] - stats[1024 + d]) * stats[1280 + d] * bnrg[d] + bnrb[d];
    if (d < 64) { int a = arch[d]; al[d] = (a == 0) ? 0.0f : ((a == 1) ? 1.0f : -1.0f); }
    __syncthreads();
    const int k = d >> 6, l = d & 63;
    float hs = 0.0f, ts = 0.0f;
    #pragma unroll
    for (int i = 0; i < 4; ++i) {
        float ri = re[i * 64 + l];
        #pragma unroll
        for (int j = 0; j < 4; ++j) {
            hs += al[i * 16 + j * 4 + k] * ri * te[j * 64 + l];  // alpha[i,j,k]
            ts += al[i * 16 + k * 4 + j] * ri * he[j * 64 + l];  // alpha[i,k,j]
        }
    }
    hvec[b * NDIM + d] = (_Float16)hs;
    tvec[b * NDIM + d] = (_Float16)ts;
}

// ------------------------------------------------------------------
// Kernel 3: f32 -> f16 copy of ent_w (vectorized)
// ------------------------------------------------------------------
__global__ void k_cvt(const float* __restrict__ src, _Float16* __restrict__ dst) {
    int i = blockIdx.x * blockDim.x + threadIdx.x;
    const int stride = gridDim.x * blockDim.x;
    const int n4 = N_ENT * NDIM / 4;
    for (; i < n4; i += stride) {
        float4 v = ((const float4*)src)[i];
        v4h h;
        h[0] = (_Float16)v.x; h[1] = (_Float16)v.y;
        h[2] = (_Float16)v.z; h[3] = (_Float16)v.w;
        ((v4h*)dst)[i] = h;
    }
}

// ------------------------------------------------------------------
// Kernel 4: WMMA GEMM + per-lane online softmax stats over an entity
// sub-range. grid(64, 2, ESPLIT), 128 threads (4 waves).
// Double-buffered B tiles; no cross-lane traffic in the hot loop.
// ------------------------------------------------------------------
__global__ __launch_bounds__(128) void k_pass1(const _Float16* __restrict__ entwh,
                                               const _Float16* __restrict__ hvec,
                                               const _Float16* __restrict__ tvec,
                                               float* __restrict__ rowM,
                                               float* __restrict__ rowS) {
    const int rowbase = blockIdx.x * 16;
    const int yt = blockIdx.y;
    const int z  = blockIdx.z;
    const int zs = (z * NTILES) / ESPLIT;
    const int ze = ((z + 1) * NTILES) / ESPLIT;
    const _Float16* vec = yt ? tvec : hvec;
    const int lane = threadIdx.x & 31, wv = threadIdx.x >> 5;
    const int m = lane & 15, hi = lane >> 4;
    const int n = lane & 15, koff = hi * 16;

    // Preload A fragments (16 batch rows x full K=256); resident all loop.
    v16h A[8];
    const _Float16* vrow = vec + (rowbase + m) * NDIM;
    #pragma unroll
    for (int kc = 0; kc < 8; ++kc) {
        v8h a0 = *(const v8h*)(vrow + kc * 32 + hi * 16);
        v8h a1 = *(const v8h*)(vrow + kc * 32 + hi * 16 + 8);
        A[kc] = cat8(a0, a1);
    }

    float mr[8], sr[8];
    #pragma unroll
    for (int g = 0; g < 8; ++g) { mr[g] = -3.402823466e38f; sr[g] = 0.0f; }

    #define BPTR(T) (entwh + (size_t)((T) * 16 + n) * NDIM + koff)
    #define PF(T)   { int tp_ = (T) + 8 < NTILES ? (T) + 8 : (NTILES - 1); \
                      __builtin_prefetch((const void*)BPTR(tp_), 0, 3); }

    v16h Ba[8], Bb[8];
    int t = zs + wv;
    if (t < ze) {
        loadB(BPTR(t), Ba);
        while (true) {
            int t2 = t + 4;
            if (t2 < ze) { PF(t2); loadB(BPTR(t2), Bb); }
            { v8f acc = mmTile(A, Ba); smUpdate(acc, mr, sr); }
            if (t2 >= ze) break;
            int t3 = t2 + 4;
            if (t3 < ze) { PF(t3); loadB(BPTR(t3), Ba); }
            { v8f acc = mmTile(A, Bb); smUpdate(acc, mr, sr); }
            if (t3 >= ze) break;
            t = t3;
        }
    }
    #undef PF
    #undef BPTR

    // One-time cross-lane (16-lane half) reduction of per-lane stats.
    #pragma unroll
    for (int g = 0; g < 8; ++g) {
        float M = mr[g];
        #pragma unroll
        for (int msk = 1; msk < 16; msk <<= 1) M = fmaxf(M, __shfl_xor(M, msk, 32));
        float S = sr[g] * __expf(mr[g] - M);
        #pragma unroll
        for (int msk = 1; msk < 16; msk <<= 1) S += __shfl_xor(S, msk, 32);
        mr[g] = M; sr[g] = S;
    }

    __shared__ float lm[4][16], ls[4][16];
    if ((lane & 15) == 0) {
        int off = hi * 8;
        #pragma unroll
        for (int g = 0; g < 8; ++g) { lm[wv][off + g] = mr[g]; ls[wv][off + g] = sr[g]; }
    }
    __syncthreads();
    const int tid = threadIdx.x;
    if (tid < 16) {
        float M = -3.402823466e38f;
        for (int w = 0; w < 4; ++w) M = fmaxf(M, lm[w][tid]);
        float S = 0.0f;
        for (int w = 0; w < 4; ++w) S += ls[w][tid] * __expf(lm[w][tid] - M);
        rowM[(yt * ESPLIT + z) * BB + rowbase + tid] = M;
        rowS[(yt * ESPLIT + z) * BB + rowbase + tid] = S;
    }
}

// ------------------------------------------------------------------
// Kernel 5: combine per-split stats into per-row LSE.
// ------------------------------------------------------------------
__global__ void k_combine(const float* __restrict__ rowM,
                          const float* __restrict__ rowS,
                          float* __restrict__ lse) {
    int idx = blockIdx.x * blockDim.x + threadIdx.x;   // 0 .. 2*BB-1
    if (idx >= 2 * BB) return;
    int yt = idx >> 10, b = idx & (BB - 1);
    float M = -3.402823466e38f;
    for (int z = 0; z < ESPLIT; ++z)
        M = fmaxf(M, rowM[(yt * ESPLIT + z) * BB + b]);
    float S = 0.0f;
    for (int z = 0; z < ESPLIT; ++z)
        S += rowS[(yt * ESPLIT + z) * BB + b] * __expf(rowM[(yt * ESPLIT + z) * BB + b] - M);
    lse[idx] = M + __logf(S);
}

// ------------------------------------------------------------------
// Kernel 6: recompute logits with WMMA, accumulate BCE terms.
// ------------------------------------------------------------------
__global__ __launch_bounds__(128) void k_pass2(const int* __restrict__ facts,
                                               const _Float16* __restrict__ entwh,
                                               const _Float16* __restrict__ hvec,
                                               const _Float16* __restrict__ tvec,
                                               const float* __restrict__ lseg,
                                               float* __restrict__ accum) {
    const int rowbase = blockIdx.x * 16;
    const int yt = blockIdx.y;
    const int z  = blockIdx.z;
    const int zs = (z * NTILES) / ESPLIT;
    const int ze = ((z + 1) * NTILES) / ESPLIT;
    const _Float16* vec = yt ? tvec : hvec;
    const int lane = threadIdx.x & 31, wv = threadIdx.x >> 5;
    const int m = lane & 15, hi = lane >> 4;
    const int n = lane & 15, koff = hi * 16;

    __shared__ float lse[16];
    __shared__ int   lbl[16];
    __shared__ float red[128];
    if (threadIdx.x < 16) {
        int b = rowbase + threadIdx.x;
        lse[threadIdx.x] = lseg[yt * BB + b];
        lbl[threadIdx.x] = facts[3 * b + yt];   // yt=0 -> h label, yt=1 -> t label
    }
    __syncthreads();

    v16h A[8];
    const _Float16* vrow = vec + (rowbase + m) * NDIM;
    #pragma unroll
    for (int kc = 0; kc < 8; ++kc) {
        v8h a0 = *(const v8h*)(vrow + kc * 32 + hi * 16);
        v8h a1 = *(const v8h*)(vrow + kc * 32 + hi * 16 + 8);
        A[kc] = cat8(a0, a1);
    }

    // This lane's LSE per accumulator slot (row = g + hi*8) and label.
    float lseL[8]; int lblL[8];
    #pragma unroll
    for (int g = 0; g < 8; ++g) { lseL[g] = lse[g + hi * 8]; lblL[g] = lbl[g + hi * 8]; }

    float local = 0.0f;

    #define BPTR(T) (entwh + (size_t)((T) * 16 + n) * NDIM + koff)
    #define PF(T)   { int tp_ = (T) + 8 < NTILES ? (T) + 8 : (NTILES - 1); \
                      __builtin_prefetch((const void*)BPTR(tp_), 0, 3); }
    #define BCE(ACC, TI)                                                          \
        {                                                                         \
            int e_ = (TI) * 16 + n;                                               \
            _Pragma("unroll")                                                     \
            for (int g = 0; g < 8; ++g) {                                         \
                float lp = (ACC)[g] - lseL[g];                                    \
                float p  = __expf(lp);                                            \
                float term = (e_ == lblL[g]) ? fmaxf(lp, -100.0f)                 \
                                             : fmaxf(log1pf(-p), -100.0f);        \
                local += term;                                                    \
            }                                                                     \
        }

    v16h Ba[8], Bb[8];
    int t = zs + wv;
    if (t < ze) {
        loadB(BPTR(t), Ba);
        while (true) {
            int t2 = t + 4;
            if (t2 < ze) { PF(t2); loadB(BPTR(t2), Bb); }
            { v8f acc = mmTile(A, Ba); BCE(acc, t); }
            if (t2 >= ze) break;
            int t3 = t2 + 4;
            if (t3 < ze) { PF(t3); loadB(BPTR(t3), Ba); }
            { v8f acc = mmTile(A, Bb); BCE(acc, t2); }
            if (t3 >= ze) break;
            t = t3;
        }
    }
    #undef BCE
    #undef PF
    #undef BPTR

    red[threadIdx.x] = local;
    __syncthreads();
    for (int s = 64; s > 0; s >>= 1) {
        if (threadIdx.x < s) red[threadIdx.x] += red[threadIdx.x + s];
        __syncthreads();
    }
    if (threadIdx.x == 0) atomicAdd(accum, red[0]);
}

// ------------------------------------------------------------------
// Kernel 7: final scalar
// ------------------------------------------------------------------
__global__ void k_finish(const float* __restrict__ accum, float* __restrict__ out) {
    out[0] = -accum[0] * (1.0f / ((float)BB * (float)N_ENT));
}

// ------------------------------------------------------------------
extern "C" void kernel_launch(void* const* d_in, const int* in_sizes, int n_in,
                              void* d_out, int out_size, void* d_ws, size_t ws_size,
                              hipStream_t stream) {
    const int*   facts = (const int*)d_in[0];
    const int*   arch  = (const int*)d_in[1];
    const float* ent_w = (const float*)d_in[2];
    const float* rel_w = (const float*)d_in[3];
    const float* bneg  = (const float*)d_in[4];
    const float* bneb  = (const float*)d_in[5];
    const float* bnrg  = (const float*)d_in[6];
    const float* bnrb  = (const float*)d_in[7];
    float* out = (float*)d_out;

    char* w = (char*)d_ws;
    float*    stats = (float*)(w);                         // 1536 f32
    float*    accum = (float*)(w + 8192);                  // 1 f32
    float*    lse   = (float*)(w + 16384);                 // [2][1024]
    float*    rowM  = (float*)(w + 32768);                 // [2][ESPLIT][1024]
    float*    rowS  = (float*)(w + 32768 + 65536);         // [2][ESPLIT][1024]
    _Float16* hvec  = (_Float16*)(w + 262144);             // 1024*256 f16
    _Float16* tvec  = hvec + BB * NDIM;
    _Float16* entwh = (_Float16*)(w + (size_t)(2u << 20)); // 50000*256 f16 (25.6 MB)

    k_init<<<1, 1, 0, stream>>>(accum);
    k_bnstats<<<3, 256, 0, stream>>>(facts, ent_w, rel_w, stats);
    k_cvt<<<512, 256, 0, stream>>>(ent_w, entwh);
    k_build<<<BB, 256, 0, stream>>>(facts, arch, ent_w, rel_w,
                                    bneg, bneb, bnrg, bnrb, stats, hvec, tvec);
    k_pass1<<<dim3(BB / 16, 2, ESPLIT), 128, 0, stream>>>(entwh, hvec, tvec, rowM, rowS);
    k_combine<<<(2 * BB + 255) / 256, 256, 0, stream>>>(rowM, rowS, lse);
    k_pass2<<<dim3(BB / 16, 2, ESPLIT), 128, 0, stream>>>(facts, entwh, hvec, tvec, lse, accum);
    k_finish<<<1, 1, 0, stream>>>(accum, out);
}
